// MAR_52209622450490
// MI455X (gfx1250) — matrix operand order repaired and checked by reading, not compiled, not run on hardware
//
#include <hip/hip_runtime.h>
#include <hip/hip_bf16.h>

// ---------------------------------------------------------------------------
// OctFormer forward for MI455X (gfx1250, wave32, WMMA).
// f16 operands + f32 accumulation via v_wmma_f32_16x16x32_f16; f32 elsewhere.
// ---------------------------------------------------------------------------

typedef __attribute__((ext_vector_type(16))) _Float16 v16h;
typedef __attribute__((ext_vector_type(8)))  _Float16 v8h;
typedef __attribute__((ext_vector_type(8)))  float    v8f;
typedef __attribute__((ext_vector_type(4)))  float    v4f;

#define NTOK   16384
#define NSPLIT 4096
#define NVQ    12288
#define CDIM   256
#define HEADS  8
#define PWIN   256
#define HIDDIM 1024

// CDNA5 16-bit A fragment (16x32): lane L -> M=L%16, khalf=L/16.
// Element e -> K = (e&7) + (e>=8 ? 16 : 0) + khalf*8   (two contiguous 8-runs)
// B fragment (32x16): lane L -> N=L%16; element e -> K same as A (table order)
//   (two contiguous 8-runs when stored K-contiguous per column).
__device__ __forceinline__ v16h cat8(v8h a, v8h b) {
  return __builtin_shufflevector(a, b, 0, 1, 2, 3, 4, 5, 6, 7,
                                        8, 9, 10, 11, 12, 13, 14, 15);
}

// ---------------------------------------------------------------------------
// WMMA GEMM: Y[M,N] = act(A[M,K]@B[K,N] + bias) (+ residual)
// Block tile 256x64, 8 waves; wave w -> rows [w*32, w*32+32) x 64 cols
// (two A fragments reuse each B fragment -> 8 WMMAs per 12 b128 LDS loads).
// ---------------------------------------------------------------------------
#define GEMM_BM 256
#define GEMM_BN 64
#define GEMM_BK 32
#define APAD    40   // padded row length (halves), 16B-aligned stride

template <int ACT, bool RES, bool OF32, bool OF16>
__global__ __launch_bounds__(256) void gemm_wmma_kernel(
    const _Float16* __restrict__ A, const _Float16* __restrict__ B,
    const float* __restrict__ bias, const float* __restrict__ residual,
    float* __restrict__ Yf32, _Float16* __restrict__ Yf16,
    int M, int N, int K)
{
  __shared__ _Float16 As[GEMM_BM][APAD];  // row-major A tile (20 KB)
  __shared__ _Float16 Bt[GEMM_BN][APAD];  // transposed B tile: Bt[n][k] (5 KB)

  const int tiles_n = N / GEMM_BN;
  const int bm = blockIdx.x / tiles_n;
  const int bn = blockIdx.x % tiles_n;
  const int tid = threadIdx.x;
  const int wave = tid >> 5;
  const int lane = tid & 31;
  const int mloc = lane & 15;
  const int khalf = lane >> 4;
  const int row0 = bm * GEMM_BM;
  const int col0 = bn * GEMM_BN;

  const int bk = tid >> 3, bn0 = (tid & 7) * 8;  // B cooperative-load indices

  v8f acc[2][4];
#pragma unroll
  for (int s = 0; s < 2; ++s)
#pragma unroll
    for (int t = 0; t < 4; ++t) acc[s][t] = (v8f){};

  for (int k0 = 0; k0 < K; k0 += GEMM_BK) {
    // A tile: each thread copies one full row (32 halves) with b128 ops
    {
      const v8h* ga = (const v8h*)(A + (size_t)(row0 + tid) * K + k0);
      v8h a0 = ga[0], a1 = ga[1], a2 = ga[2], a3 = ga[3];
      *(v8h*)&As[tid][0]  = a0;
      *(v8h*)&As[tid][8]  = a1;
      *(v8h*)&As[tid][16] = a2;
      *(v8h*)&As[tid][24] = a3;
    }
    // B tile: vector global read, transposed LDS writes
    {
      v8h bv = *(const v8h*)(B + (size_t)(k0 + bk) * N + col0 + bn0);
#pragma unroll
      for (int j = 0; j < 8; ++j) Bt[bn0 + j][bk] = bv[j];
    }
    __syncthreads();

    v16h af[2];
#pragma unroll
    for (int s = 0; s < 2; ++s) {
      const int ar = wave * 32 + s * 16 + mloc;
      af[s] = cat8(*(const v8h*)&As[ar][khalf * 8],
                   *(const v8h*)&As[ar][16 + khalf * 8]);
    }
#pragma unroll
    for (int t = 0; t < 4; ++t) {
      v16h bf = cat8(*(const v8h*)&Bt[t * 16 + mloc][khalf * 8],
                     *(const v8h*)&Bt[t * 16 + mloc][16 + khalf * 8]);
      acc[0][t] = __builtin_amdgcn_wmma_f32_16x16x32_f16(
          false, af[0], false, bf, (short)0, acc[0][t], false, false);
      acc[1][t] = __builtin_amdgcn_wmma_f32_16x16x32_f16(
          false, af[1], false, bf, (short)0, acc[1][t], false, false);
    }
    __syncthreads();
  }

  // Epilogue: C/D layout -> lane: M = r + 8*khalf, N = lane%16 (per 16x16 tile)
#pragma unroll
  for (int s = 0; s < 2; ++s) {
#pragma unroll
    for (int t = 0; t < 4; ++t) {
      const int gn = col0 + t * 16 + mloc;
      const float bval = bias[gn];
#pragma unroll
      for (int r = 0; r < 8; ++r) {
        int gm = row0 + wave * 32 + s * 16 + r + 8 * khalf;
        float v = acc[s][t][r] + bval;
        if (ACT == 1) {  // tanh-approx GELU (jax.nn.gelu default)
          float u = 0.7978845608028654f * (v + 0.044715f * v * v * v);
          v = 0.5f * v * (1.0f + tanhf(u));
        }
        size_t idx = (size_t)gm * N + gn;
        if (RES) v += residual[idx];
        if (OF32) Yf32[idx] = v;
        if (OF16) Yf16[idx] = (_Float16)v;
      }
    }
  }
}

// ---------------------------------------------------------------------------
// Windowed attention, one wave per (window, head, 16-row strip).
// qkv: [NTOK, 768] f16 (q|k|v each 256 cols; head h at cols h*32..h*32+31)
// o:   [NTOK, 256] f16
// S fragments stay in registers; softmax via shfl reductions; only f16
// probabilities pass through LDS (C-layout -> A-layout conversion).
// ---------------------------------------------------------------------------
__global__ __launch_bounds__(32) void attn_kernel(
    const _Float16* __restrict__ qkv, _Float16* __restrict__ o, int dil)
{
  __shared__ _Float16 Sh[16][PWIN];       // 8 KB probabilities
  __shared__ _Float16 Vt[32][PWIN + 8];   // transposed V tile: Vt[dch][tok]

  const int bid = blockIdx.x;
  const int rb = bid & 15;
  const int head = (bid >> 4) & 7;
  const int win = bid >> 7;
  const int sw = win / dil;
  const int j = win % dil;
  const int base = sw * PWIN * dil + j;
  const int lane = threadIdx.x;
  const int mloc = lane & 15;
  const int khalf = lane >> 4;
  const float scale = 0.17677669529663687f;  // 32^-0.5

  // --- stage V transposed (each lane handles 8 token rows) ---
  for (int tt = lane; tt < PWIN; tt += 32) {
    const v8h* vr = (const v8h*)(qkv + (size_t)(base + tt * dil) * 768 +
                                 512 + head * 32);
    v8h v0 = vr[0], v1 = vr[1], v2 = vr[2], v3 = vr[3];
#pragma unroll
    for (int jj = 0; jj < 8; ++jj) {
      Vt[jj][tt]      = v0[jj];
      Vt[8 + jj][tt]  = v1[jj];
      Vt[16 + jj][tt] = v2[jj];
      Vt[24 + jj][tt] = v3[jj];
    }
  }

  // --- Q fragment: two contiguous 16B runs from global ---
  const int tokm = base + (rb * 16 + mloc) * dil;
  const _Float16* qrow = qkv + (size_t)tokm * 768 + head * 32;
  v16h aq = cat8(*(const v8h*)(qrow + khalf * 8),
                 *(const v8h*)(qrow + 16 + khalf * 8));

  // --- S = Q K^T, 16 fragments kept in registers ---
  v8f sacc[16];
#pragma unroll
  for (int nt = 0; nt < 16; ++nt) sacc[nt] = (v8f){};
#pragma unroll
  for (int nt = 0; nt < 16; ++nt) {
    const int ntok = base + (nt * 16 + mloc) * dil;
    const _Float16* krow =
        qkv + (size_t)ntok * 768 + 256 + head * 32 + 8 * khalf;
    v16h bk = cat8(*(const v8h*)krow, *(const v8h*)(krow + 16));
    sacc[nt] = __builtin_amdgcn_wmma_f32_16x16x32_f16(
        false, aq, false, bk, (short)0, sacc[nt], false, false);
  }

  // --- softmax: row r+8*khalf held by the 16 lanes sharing khalf ---
  float rinv[8];
#pragma unroll
  for (int r = 0; r < 8; ++r) {
    float mx = -1e30f;
#pragma unroll
    for (int nt = 0; nt < 16; ++nt) mx = fmaxf(mx, sacc[nt][r] * scale);
    mx = fmaxf(mx, __shfl_xor(mx, 1, 32));
    mx = fmaxf(mx, __shfl_xor(mx, 2, 32));
    mx = fmaxf(mx, __shfl_xor(mx, 4, 32));
    mx = fmaxf(mx, __shfl_xor(mx, 8, 32));
    float sum = 0.0f;
#pragma unroll
    for (int nt = 0; nt < 16; ++nt) {
      float ev = __expf(sacc[nt][r] * scale - mx);
      Sh[r + 8 * khalf][nt * 16 + mloc] = (_Float16)ev;
      sum += ev;
    }
    sum += __shfl_xor(sum, 1, 32);
    sum += __shfl_xor(sum, 2, 32);
    sum += __shfl_xor(sum, 4, 32);
    sum += __shfl_xor(sum, 8, 32);
    rinv[r] = 1.0f / sum;
  }
  __syncthreads();

  // --- O = P V  (M=16, K=256, N=32), all fragments as b128 LDS loads ---
  v8f oacc[2] = {{}, {}};
#pragma unroll
  for (int kb = 0; kb < 8; ++kb) {
    v16h as = cat8(*(const v8h*)&Sh[mloc][kb * 32 + khalf * 8],
                   *(const v8h*)&Sh[mloc][kb * 32 + 16 + khalf * 8]);
#pragma unroll
    for (int t = 0; t < 2; ++t) {
      v16h bv = cat8(*(const v8h*)&Vt[t * 16 + mloc][kb * 32 + khalf * 8],
                     *(const v8h*)&Vt[t * 16 + mloc][kb * 32 + 16 + khalf * 8]);
      oacc[t] = __builtin_amdgcn_wmma_f32_16x16x32_f16(
          false, as, false, bv, (short)0, oacc[t], false, false);
    }
  }
#pragma unroll
  for (int t = 0; t < 2; ++t)
#pragma unroll
    for (int r = 0; r < 8; ++r) {
      int tok = base + (rb * 16 + r + 8 * khalf) * dil;
      o[(size_t)tok * 256 + head * 32 + t * 16 + mloc] =
          (_Float16)(oacc[t][r] * rinv[r]);
    }
}

// ---------------------------------------------------------------------------
// LayerNorm over C=256 (one wave per row, 8 rows per block), f16 output.
// ---------------------------------------------------------------------------
__global__ __launch_bounds__(256) void ln_kernel(
    const float* __restrict__ x, const float* __restrict__ s,
    const float* __restrict__ b, _Float16* __restrict__ out)
{
  const int row = blockIdx.x * 8 + (threadIdx.x >> 5);
  const int lane = threadIdx.x & 31;
  const v4f* xr4 = (const v4f*)(x + (size_t)row * CDIM);
  v4f p0 = xr4[lane * 2], p1 = xr4[lane * 2 + 1];
  float v[8], m = 0.0f;
#pragma unroll
  for (int i = 0; i < 4; ++i) { v[i] = p0[i]; v[i + 4] = p1[i]; }
#pragma unroll
  for (int i = 0; i < 8; ++i) m += v[i];
  for (int off = 16; off; off >>= 1) m += __shfl_xor(m, off, 32);
  m *= (1.0f / CDIM);
  float var = 0.0f;
#pragma unroll
  for (int i = 0; i < 8; ++i) { float d = v[i] - m; var += d * d; }
  for (int off = 16; off; off >>= 1) var += __shfl_xor(var, off, 32);
  var *= (1.0f / CDIM);
  float rs = rsqrtf(var + 1e-5f);
  v4f s0 = *(const v4f*)(s + lane * 8), s1 = *(const v4f*)(s + lane * 8 + 4);
  v4f b0 = *(const v4f*)(b + lane * 8), b1 = *(const v4f*)(b + lane * 8 + 4);
  v8h ov;
#pragma unroll
  for (int i = 0; i < 4; ++i) {
    ov[i]     = (_Float16)((v[i] - m) * rs * s0[i] + b0[i]);
    ov[i + 4] = (_Float16)((v[i + 4] - m) * rs * s1[i] + b1[i]);
  }
  *(v8h*)(out + (size_t)row * CDIM + lane * 8) = ov;
}

// Final LN fused with batch2depth scatter: xf[d2b[row]] = LN(xb[row])
__global__ __launch_bounds__(256) void scatter_ln_kernel(
    const float* __restrict__ xbuf, const float* __restrict__ s,
    const float* __restrict__ b, const int* __restrict__ d2b,
    float* __restrict__ xf, _Float16* __restrict__ xh)
{
  const int row = blockIdx.x * 8 + (threadIdx.x >> 5);
  const int lane = threadIdx.x & 31;
  const v4f* xr4 = (const v4f*)(xbuf + (size_t)row * CDIM);
  v4f p0 = xr4[lane * 2], p1 = xr4[lane * 2 + 1];
  float v[8], m = 0.0f;
#pragma unroll
  for (int i = 0; i < 4; ++i) { v[i] = p0[i]; v[i + 4] = p1[i]; }
#pragma unroll
  for (int i = 0; i < 8; ++i) m += v[i];
  for (int off = 16; off; off >>= 1) m += __shfl_xor(m, off, 32);
  m *= (1.0f / CDIM);
  float var = 0.0f;
#pragma unroll
  for (int i = 0; i < 8; ++i) { float d = v[i] - m; var += d * d; }
  for (int off = 16; off; off >>= 1) var += __shfl_xor(var, off, 32);
  var *= (1.0f / CDIM);
  float rs = rsqrtf(var + 1e-5f);
  const int orow = d2b[row];
  v4f s0 = *(const v4f*)(s + lane * 8), s1 = *(const v4f*)(s + lane * 8 + 4);
  v4f b0 = *(const v4f*)(b + lane * 8), b1 = *(const v4f*)(b + lane * 8 + 4);
  v4f o0, o1;
  v8h ov;
#pragma unroll
  for (int i = 0; i < 4; ++i) {
    o0[i] = (v[i] - m) * rs * s0[i] + b0[i];
    o1[i] = (v[i + 4] - m) * rs * s1[i] + b1[i];
    ov[i] = (_Float16)o0[i];
    ov[i + 4] = (_Float16)o1[i];
  }
  *(v4f*)(xf + (size_t)orow * CDIM + lane * 8)     = o0;
  *(v4f*)(xf + (size_t)orow * CDIM + lane * 8 + 4) = o1;
  *(v8h*)(xh + (size_t)orow * CDIM + lane * 8)     = ov;
}

// ---------------------------------------------------------------------------
// Embedding + mask/cond + d2b gather + sinusoidal position embedding.
// ---------------------------------------------------------------------------
__global__ void embed_kernel(
    const int* __restrict__ split, const float* __restrict__ zq,
    const int* __restrict__ category, const int* __restrict__ batch_id,
    const int* __restrict__ mask, const int* __restrict__ d2b,
    const float* __restrict__ split_emb, const float* __restrict__ class_emb,
    const float* __restrict__ vq_proj_w, const float* __restrict__ vq_proj_b,
    float* __restrict__ xb)
{
  const int idx = blockIdx.x * 256 + threadIdx.x;
  if (idx >= NTOK * CDIM) return;
  const int i = idx >> 8;
  const int c = idx & 255;
  const int jtok = d2b[i];

  float val;
  if (mask[jtok]) {
    val = class_emb[category[batch_id[jtok]] * CDIM + c];
  } else if (jtok < NSPLIT) {
    val = split_emb[split[jtok] * CDIM + c];
  } else {
    const int r = jtok - NSPLIT;
    float acc = vq_proj_b[c];
    for (int k = 0; k < 32; ++k) acc += zq[r * 32 + k] * vq_proj_w[k * CDIM + c];
    val = acc;
  }
  const int half = c & 127;
  const float freq = __expf(-logf(10000.0f) * (float)half * (1.0f / 128.0f));
  const float ang = (float)i * freq;
  const float pe = (c < 128) ? sinf(ang) : cosf(ang);
  xb[idx] = val + pe;
}

// ---------------------------------------------------------------------------
// Split head + masked CE. accum: [0]=ce_s, [1]=ms, [2]=ce_v, [3]=mv
// ---------------------------------------------------------------------------
__global__ void split_ce_kernel(
    const float* __restrict__ xf, const float* __restrict__ split_w,
    const float* __restrict__ split_b, const int* __restrict__ split,
    const int* __restrict__ mask, float* __restrict__ accum)
{
  const int r = blockIdx.x * 256 + threadIdx.x;
  if (r >= NSPLIT) return;
  float l0 = split_b[0], l1 = split_b[1];
  const float* xr = xf + (size_t)r * CDIM;
  for (int k = 0; k < CDIM; ++k) {
    float xv = xr[k];
    l0 += xv * split_w[k * 2 + 0];
    l1 += xv * split_w[k * 2 + 1];
  }
  float mx = fmaxf(l0, l1);
  float lse = mx + logf(__expf(l0 - mx) + __expf(l1 - mx));
  float ce = lse - (split[r] ? l1 : l0);
  float ms = mask[r] ? 1.0f : 0.0f;
  atomicAdd(&accum[0], ce * ms);
  atomicAdd(&accum[1], ms);
}

// ---------------------------------------------------------------------------
// VQ head + grouped masked CE, one wave per VQ token.
// Uses transposed f16 weights (vq_wT[col][k]) for contiguous dot products.
// ---------------------------------------------------------------------------
__global__ __launch_bounds__(256) void vq_ce_kernel(
    const _Float16* __restrict__ xh, const _Float16* __restrict__ vq_wT,
    const float* __restrict__ vq_b, const int* __restrict__ targets,
    const int* __restrict__ mask, float* __restrict__ accum)
{
  __shared__ _Float16 rows[8][CDIM];
  const int wave = threadIdx.x >> 5, lane = threadIdx.x & 31;
  const int r = blockIdx.x * 8 + wave;
  const _Float16* xr = xh + (size_t)(NSPLIT + r) * CDIM;
  *(v8h*)&rows[wave][lane * 8] = *(const v8h*)&xr[lane * 8];

  float mg[4], sg[4], tg[4];
#pragma unroll
  for (int g = 0; g < 4; ++g) { mg[g] = -1e30f; sg[g] = 0.0f; tg[g] = 0.0f; }

  for (int ii = 0; ii < 32; ++ii) {
    const int c = lane + 32 * ii;
    const int g = ii >> 3;
    const _Float16* wcol = vq_wT + (size_t)c * CDIM;
    float acc = vq_b[c];
    for (int k = 0; k < CDIM; k += 8) {
      v8h rv = *(const v8h*)&rows[wave][k];
      v8h wv = *(const v8h*)&wcol[k];
#pragma unroll
      for (int jj = 0; jj < 8; ++jj) acc += (float)rv[jj] * (float)wv[jj];
    }
    float nm = fmaxf(mg[g], acc);
    sg[g] = sg[g] * __expf(mg[g] - nm) + __expf(acc - nm);
    mg[g] = nm;
    if ((c & 255) == targets[r * 4 + g]) tg[g] = acc;
  }

  float ce = 0.0f;
#pragma unroll
  for (int g = 0; g < 4; ++g) {
    float m = mg[g], s = sg[g];
    for (int off = 16; off; off >>= 1) {
      float mo = __shfl_xor(m, off, 32);
      float so = __shfl_xor(s, off, 32);
      float nm = fmaxf(m, mo);
      s = s * __expf(m - nm) + so * __expf(mo - nm);
      m = nm;
    }
    float t = tg[g];
    for (int off = 16; off; off >>= 1) t += __shfl_xor(t, off, 32);
    ce += (m + logf(s)) - t;
  }
  ce *= 0.25f;
  if (lane == 0) {
    float mv = mask[NSPLIT + r] ? 1.0f : 0.0f;
    atomicAdd(&accum[2], ce * mv);
    atomicAdd(&accum[3], mv);
  }
}

__global__ void f32_to_f16_kernel(const float* __restrict__ in,
                                  _Float16* __restrict__ out, int n) {
  int i = blockIdx.x * 256 + threadIdx.x;
  if (i < n) out[i] = (_Float16)in[i];
}

// vq_w [256][1024] f32 -> vq_wT [1024][256] f16
__global__ void transpose_to_f16_kernel(const float* __restrict__ in,
                                        _Float16* __restrict__ out) {
  int idx = blockIdx.x * 256 + threadIdx.x;
  int k = idx >> 10, c = idx & 1023;
  out[(size_t)c * CDIM + k] = (_Float16)in[idx];
}

__global__ void zero_kernel(float* a) {
  if (threadIdx.x < 4) a[threadIdx.x] = 0.0f;
}

__global__ void finalize_kernel(const float* __restrict__ accum,
                                float* __restrict__ out) {
  if (threadIdx.x == 0) {
    out[0] = accum[0] / fmaxf(accum[1], 1.0f);
    out[1] = accum[2] / fmaxf(accum[3], 1.0f);
  }
}

// ---------------------------------------------------------------------------
extern "C" void kernel_launch(void* const* d_in, const int* in_sizes, int n_in,
                              void* d_out, int out_size, void* d_ws, size_t ws_size,
                              hipStream_t stream) {
  (void)in_sizes; (void)n_in; (void)out_size; (void)ws_size;

  const int*   split      = (const int*)  d_in[0];
  const float* zq         = (const float*)d_in[1];
  const int*   targets_vq = (const int*)  d_in[2];
  const int*   category   = (const int*)  d_in[3];
  const int*   batch_id   = (const int*)  d_in[4];
  const int*   mask       = (const int*)  d_in[5];
  const int*   d2b        = (const int*)  d_in[6];
  const float* split_emb  = (const float*)d_in[7];
  const float* class_emb  = (const float*)d_in[8];
  const float* vq_proj_w  = (const float*)d_in[9];
  const float* vq_proj_b  = (const float*)d_in[10];
  const float* ln1_s      = (const float*)d_in[11];
  const float* ln1_b      = (const float*)d_in[12];
  const float* qkv_w      = (const float*)d_in[13];
  const float* qkv_b      = (const float*)d_in[14];
  const float* attn_w     = (const float*)d_in[15];
  const float* attn_b     = (const float*)d_in[16];
  const float* ln2_s      = (const float*)d_in[17];
  const float* ln2_b      = (const float*)d_in[18];
  const float* fc1_w      = (const float*)d_in[19];
  const float* fc1_b      = (const float*)d_in[20];
  const float* fc2_w      = (const float*)d_in[21];
  const float* fc2_b      = (const float*)d_in[22];
  const float* lnx_s      = (const float*)d_in[23];
  const float* lnx_b      = (const float*)d_in[24];
  const float* split_w    = (const float*)d_in[25];
  const float* split_b    = (const float*)d_in[26];
  const float* vq_w       = (const float*)d_in[27];
  const float* vq_b       = (const float*)d_in[28];

  char* ws = (char*)d_ws;
  size_t off = 0;
  auto alloc = [&](size_t bytes) -> void* {
    void* p = ws + off;
    off = (off + bytes + 255) & ~(size_t)255;
    return p;
  };

  float*     accum    = (float*)    alloc(4 * sizeof(float));
  _Float16*  qkv_w16  = (_Float16*) alloc((size_t)4 * 256 * 768 * 2);
  _Float16*  attn_w16 = (_Float16*) alloc((size_t)4 * 256 * 256 * 2);
  _Float16*  fc1_w16  = (_Float16*) alloc((size_t)4 * 256 * 1024 * 2);
  _Float16*  fc2_w16  = (_Float16*) alloc((size_t)4 * 1024 * 256 * 2);
  _Float16*  vq_wT    = (_Float16*) alloc((size_t)1024 * 256 * 2);
  float*     xb       = (float*)    alloc((size_t)NTOK * CDIM * 4);
  _Float16*  h16      = (_Float16*) alloc((size_t)NTOK * CDIM * 2);
  _Float16*  m16      = (_Float16*) alloc((size_t)NTOK * HIDDIM * 2); // qkv & MLP
  _Float16*  o16      = (_Float16*) alloc((size_t)NTOK * CDIM * 2);
  float*     xf32     = (float*)    alloc((size_t)NTOK * CDIM * 4);
  _Float16*  xf16     = (_Float16*) alloc((size_t)NTOK * CDIM * 2);

  zero_kernel<<<1, 32, 0, stream>>>(accum);

  f32_to_f16_kernel<<<(4*256*768  + 255)/256, 256, 0, stream>>>(qkv_w,  qkv_w16,  4*256*768);
  f32_to_f16_kernel<<<(4*256*256  + 255)/256, 256, 0, stream>>>(attn_w, attn_w16, 4*256*256);
  f32_to_f16_kernel<<<(4*256*1024 + 255)/256, 256, 0, stream>>>(fc1_w,  fc1_w16,  4*256*1024);
  f32_to_f16_kernel<<<(4*1024*256 + 255)/256, 256, 0, stream>>>(fc2_w,  fc2_w16,  4*1024*256);
  transpose_to_f16_kernel<<<(256*1024)/256, 256, 0, stream>>>(vq_w, vq_wT);

  embed_kernel<<<NTOK * CDIM / 256, 256, 0, stream>>>(
      split, zq, category, batch_id, mask, d2b,
      split_emb, class_emb, vq_proj_w, vq_proj_b, xb);

  for (int l = 0; l < 4; ++l) {
    const int dil = (l & 1) ? 2 : 1;
    ln_kernel<<<NTOK / 8, 256, 0, stream>>>(xb, ln1_s + l*256, ln1_b + l*256, h16);
    // qkv = h @ W + b  -> f16 out
    gemm_wmma_kernel<0, false, false, true>
        <<<(NTOK/GEMM_BM) * (768/GEMM_BN), 256, 0, stream>>>(
        h16, qkv_w16 + (size_t)l*256*768, qkv_b + l*768,
        nullptr, nullptr, m16, NTOK, 768, 256);
    attn_kernel<<<64 * HEADS * 16, 32, 0, stream>>>(m16, o16, dil);
    // xb += o @ W + b  -> f32 out with residual
    gemm_wmma_kernel<0, true, true, false>
        <<<(NTOK/GEMM_BM) * (256/GEMM_BN), 256, 0, stream>>>(
        o16, attn_w16 + (size_t)l*256*256, attn_b + l*256,
        xb, xb, nullptr, NTOK, 256, 256);
    ln_kernel<<<NTOK / 8, 256, 0, stream>>>(xb, ln2_s + l*256, ln2_b + l*256, h16);
    // m = gelu(h @ W + b) -> f16 out
    gemm_wmma_kernel<1, false, false, true>
        <<<(NTOK/GEMM_BM) * (HIDDIM/GEMM_BN), 256, 0, stream>>>(
        h16, fc1_w16 + (size_t)l*256*1024, fc1_b + l*1024,
        nullptr, nullptr, m16, NTOK, HIDDIM, 256);
    // xb += m @ W + b -> f32 out with residual
    gemm_wmma_kernel<0, true, true, false>
        <<<(NTOK/GEMM_BM) * (256/GEMM_BN), 256, 0, stream>>>(
        m16, fc2_w16 + (size_t)l*1024*256, fc2_b + l*256,
        xb, xb, nullptr, NTOK, 256, HIDDIM);
  }

  scatter_ln_kernel<<<NTOK / 8, 256, 0, stream>>>(xb, lnx_s, lnx_b, d2b, xf32, xf16);

  split_ce_kernel<<<NSPLIT / 256, 256, 0, stream>>>(
      xf32, split_w, split_b, split, mask, accum);
  vq_ce_kernel<<<NVQ / 8, 256, 0, stream>>>(
      xf16, vq_wT, vq_b, targets_vq, mask, accum);

  finalize_kernel<<<1, 32, 0, stream>>>(accum, (float*)d_out);
}
